// TreeGRUDiscriminator_26328149525043
// MI455X (gfx1250) — matrix operand order repaired and checked by reading, not compiled, not run on hardware
//
#include <hip/hip_runtime.h>
#include <hip/hip_bf16.h>
#include <math.h>

// ---------------------------------------------------------------------------
// TreeGRU discriminator for MI455X (gfx1250, wave32, WMMA).
//  N nodes, F=64 features, H=64 hidden, 3*H=192 gates, T=4 frames, 2 layers.
// ---------------------------------------------------------------------------

#define F   64
#define G3  192

typedef _Float16 v16h __attribute__((ext_vector_type(16)));
typedef _Float16 v8h  __attribute__((ext_vector_type(8)));
typedef float    v8f  __attribute__((ext_vector_type(8)));

union AFrag { v16h v; v8h h8[2]; };

// A-matrix fragment (16x32 f16, row-major source [rows][64]).
// ISA 7.12.2: lanes 0-15 hold K=0..7 (VGPR0-3) and K=16..23 (VGPR4-7);
// lanes 16-31 hold K=8..15 and K=24..31.  -> two contiguous 16B loads.
__device__ __forceinline__ v16h load_a_frag(const _Float16* base, int row,
                                            int kbase, int lane) {
  const _Float16* p = base + (size_t)row * F + kbase + ((lane & 16) ? 8 : 0);
  AFrag f;
  f.h8[0] = *(const v8h*)(p);
  f.h8[1] = *(const v8h*)(p + 16);
  return f.v;
}

// B-matrix fragment (32x16 f16). Source is W row-major [192][64]; B[k][j]=W[j][k],
// so lane's column j selects a W row. Same K order as A: lanes 0-15 hold
// K=0..7 and K=16..23, lanes 16-31 hold K=8..15 and K=24..31. -> two 16B loads.
__device__ __forceinline__ v16h load_b_frag(const _Float16* w, int jrow,
                                            int kbase, int lane) {
  const _Float16* p = w + (size_t)jrow * F + kbase + ((lane & 16) ? 8 : 0);
  AFrag f;
  f.h8[0] = *(const v8h*)(p);
  f.h8[1] = *(const v8h*)(p + 16);
  return f.v;
}

__device__ __forceinline__ v8f wmma_f16(v16h a, v16h b, v8f c) {
  return __builtin_amdgcn_wmma_f32_16x16x32_f16(false, a, false, b,
                                                (short)0, c, false, false);
}

__device__ __forceinline__ v8f splat8(float x) {
  v8f v = {x, x, x, x, x, x, x, x};
  return v;
}

__device__ __forceinline__ float sigmoidf_(float x) {
  return 1.0f / (1.0f + expf(-x));
}

// ------------------------------- small kernels -----------------------------

__global__ void k_f32_to_f16(const float* __restrict__ in,
                             _Float16* __restrict__ out, int n) {
  int i = blockIdx.x * blockDim.x + threadIdx.x;
  if (i < n) out[i] = (_Float16)in[i];
}

__global__ void k_count(const int* __restrict__ dst, float* cnt, int E) {
  int e = blockIdx.x * blockDim.x + threadIdx.x;
  if (e < E) atomicAdd(&cnt[dst[e]], 1.0f);
}

__global__ void k_invert(float* c, int n) {
  int i = blockIdx.x * blockDim.x + threadIdx.x;
  if (i < n) c[i] = 1.0f / fmaxf(c[i], 1.0f);
}

// thread per (edge, 4-feature chunk): 16 consecutive threads stream one 256B row.
__global__ void k_scatter(const float* __restrict__ h, const int* __restrict__ src,
                          const int* __restrict__ dst, float* acc, int E) {
  int t = blockIdx.x * blockDim.x + threadIdx.x;
  if (t >= (E << 4)) return;
  int e = t >> 4;
  int c = (t & 15) << 2;
  int s = src[e];
  int d = dst[e];
  const float4 v = *(const float4*)(h + (size_t)s * F + c);
  float* a = acc + (size_t)d * F + c;
  atomicAdd(a + 0, v.x);
  atomicAdd(a + 1, v.y);
  atomicAdd(a + 2, v.z);
  atomicAdd(a + 3, v.w);
}

// scale by 1/deg (in-place on f32 frame) + produce f16 copy for WMMA A-operand.
__global__ void k_scale_convert(const float* in, const float* __restrict__ invc,
                                float* out32, _Float16* __restrict__ out16, int n) {
  int i = blockIdx.x * blockDim.x + threadIdx.x;
  if (i < n) {
    float v = in[i] * invc[i >> 6];
    out32[i] = v;
    out16[i] = (_Float16)v;
  }
}

// ------------------------------- GRU step ----------------------------------
// One wave per 16-node tile. gi = x @ W_ih^T + b_ih, gh = h @ W_hh^T + b_hh,
// torch gate order r,z,n. f16 WMMA operands, f32 accumulate + f32 gate math.
// In-place hidden update (hprev* may alias hout*): each (node,feature) is
// read and then written by the owning wave within the same jt iteration.
__global__ void __launch_bounds__(256)
k_gru_step(const _Float16* __restrict__ x16,
           const _Float16* __restrict__ wih, const _Float16* __restrict__ whh,
           const float* __restrict__ bih, const float* __restrict__ bhh,
           const _Float16* hprev16, const float* hprev32,
           float* hout32, _Float16* hout16,
           int nNodes, int firstStep) {
  int lane = threadIdx.x & 31;
  int wave = blockIdx.x * (blockDim.x >> 5) + (threadIdx.x >> 5);
  int row0 = wave * 16;
  if (row0 >= nNodes) return;  // wave-uniform exit keeps EXEC full for WMMA

  int arow = row0 + (lane & 15);
  if (arow >= nNodes) arow = nNodes - 1;  // clamp loads; stores predicated

  v16h ax0 = load_a_frag(x16, arow, 0, lane);
  v16h ax1 = load_a_frag(x16, arow, 32, lane);
  v16h ah0 = {};
  v16h ah1 = {};
  if (!firstStep) {
    ah0 = load_a_frag(hprev16, arow, 0, lane);
    ah1 = load_a_frag(hprev16, arow, 32, lane);
  }

  const int nlo = lane & 15;
  const int mbase = (lane & 16) ? 8 : 0;

#pragma unroll
  for (int jt = 0; jt < 4; ++jt) {
    const int f = jt * 16 + nlo;  // output feature 0..63 for this lane

    v8f ir = splat8(bih[f]);
    v8f iz = splat8(bih[64 + f]);
    v8f in_ = splat8(bih[128 + f]);
    v8f hr = splat8(bhh[f]);
    v8f hz = splat8(bhh[64 + f]);
    v8f hn = splat8(bhh[128 + f]);

    ir = wmma_f16(ax0, load_b_frag(wih, f, 0, lane), ir);
    ir = wmma_f16(ax1, load_b_frag(wih, f, 32, lane), ir);
    iz = wmma_f16(ax0, load_b_frag(wih, 64 + f, 0, lane), iz);
    iz = wmma_f16(ax1, load_b_frag(wih, 64 + f, 32, lane), iz);
    in_ = wmma_f16(ax0, load_b_frag(wih, 128 + f, 0, lane), in_);
    in_ = wmma_f16(ax1, load_b_frag(wih, 128 + f, 32, lane), in_);

    if (!firstStep) {
      hr = wmma_f16(ah0, load_b_frag(whh, f, 0, lane), hr);
      hr = wmma_f16(ah1, load_b_frag(whh, f, 32, lane), hr);
      hz = wmma_f16(ah0, load_b_frag(whh, 64 + f, 0, lane), hz);
      hz = wmma_f16(ah1, load_b_frag(whh, 64 + f, 32, lane), hz);
      hn = wmma_f16(ah0, load_b_frag(whh, 128 + f, 0, lane), hn);
      hn = wmma_f16(ah1, load_b_frag(whh, 128 + f, 32, lane), hn);
    }

#pragma unroll
    for (int v = 0; v < 8; ++v) {
      int node = row0 + mbase + v;
      if (node >= nNodes) continue;
      float r = sigmoidf_(ir[v] + hr[v]);
      float zt = sigmoidf_(iz[v] + hz[v]);
      float nn = tanhf(in_[v] + r * hn[v]);
      float hp = firstStep ? 0.0f : hprev32[(size_t)node * F + f];
      float hv = (1.0f - zt) * nn + zt * hp;
      hout32[(size_t)node * F + f] = hv;
      hout16[(size_t)node * F + f] = (_Float16)hv;
    }
  }
}

// out[n] = tanh(h[n,:]) . W_out + b_out
__global__ void k_head(const float* __restrict__ h, const float* __restrict__ wout,
                       const float* __restrict__ bout, float* __restrict__ out,
                       int n) {
  int i = blockIdx.x * blockDim.x + threadIdx.x;
  if (i >= n) return;
  float s = 0.0f;
#pragma unroll
  for (int f2 = 0; f2 < F; ++f2) s += tanhf(h[(size_t)i * F + f2]) * wout[f2];
  out[i] = s + bout[0];
}

// ------------------------------- launcher ----------------------------------

extern "C" void kernel_launch(void* const* d_in, const int* in_sizes, int n_in,
                              void* d_out, int out_size, void* d_ws, size_t ws_size,
                              hipStream_t stream) {
  (void)n_in; (void)out_size; (void)ws_size;

  const float* z = (const float*)d_in[0];
  const int* eidx = (const int*)d_in[1];
  const float* W_ih0 = (const float*)d_in[2];
  const float* W_hh0 = (const float*)d_in[3];
  const float* b_ih0 = (const float*)d_in[4];
  const float* b_hh0 = (const float*)d_in[5];
  const float* W_ih1 = (const float*)d_in[6];
  const float* W_hh1 = (const float*)d_in[7];
  const float* b_ih1 = (const float*)d_in[8];
  const float* b_hh1 = (const float*)d_in[9];
  const float* W_out = (const float*)d_in[10];
  const float* b_out = (const float*)d_in[11];
  float* out = (float*)d_out;

  const int N = in_sizes[0] / F;
  const int E = in_sizes[1] / 2;
  const int* srcI = eidx;
  const int* dstI = eidx + E;

  // workspace carve-up, 256B aligned
  char* p = (char*)d_ws;
  auto take = [&p](size_t bytes) -> char* {
    char* q = p;
    p += (bytes + 255) & ~(size_t)255;
    return q;
  };
  float*    invcnt = (float*)take((size_t)N * 4);
  float*    aggA   = (float*)take((size_t)N * F * 4);
  float*    aggB   = (float*)take((size_t)N * F * 4);
  _Float16* x16    = (_Float16*)take((size_t)N * F * 2);
  float*    h0_32  = (float*)take((size_t)N * F * 4);
  _Float16* h0_16  = (_Float16*)take((size_t)N * F * 2);
  float*    h1_32  = (float*)take((size_t)N * F * 4);
  _Float16* h1_16  = (_Float16*)take((size_t)N * F * 2);
  _Float16* wih0   = (_Float16*)take((size_t)G3 * F * 2);
  _Float16* whh0   = (_Float16*)take((size_t)G3 * F * 2);
  _Float16* wih1   = (_Float16*)take((size_t)G3 * F * 2);
  _Float16* whh1   = (_Float16*)take((size_t)G3 * F * 2);

  const int TB = 256;
  const int wn = G3 * F;
  k_f32_to_f16<<<(wn + TB - 1) / TB, TB, 0, stream>>>(W_ih0, wih0, wn);
  k_f32_to_f16<<<(wn + TB - 1) / TB, TB, 0, stream>>>(W_hh0, whh0, wn);
  k_f32_to_f16<<<(wn + TB - 1) / TB, TB, 0, stream>>>(W_ih1, wih1, wn);
  k_f32_to_f16<<<(wn + TB - 1) / TB, TB, 0, stream>>>(W_hh1, whh1, wn);

  // degrees -> reciprocal (counts are exact in f32 for E <= 2^24 per node)
  hipMemsetAsync(invcnt, 0, (size_t)N * 4, stream);
  k_count<<<(E + TB - 1) / TB, TB, 0, stream>>>(dstI, invcnt, E);
  k_invert<<<(N + TB - 1) / TB, TB, 0, stream>>>(invcnt, N);

  const int nf = N * F;
  const int gruBlocks = ((N + 15) / 16 + 7) / 8;  // 8 waves (tiles) per block
  const float* cur = z;

  for (int t = 0; t < 4; ++t) {
    if (t == 0) {
      k_f32_to_f16<<<(nf + TB - 1) / TB, TB, 0, stream>>>(z, x16, nf);
    } else {
      float* acc = (t & 1) ? aggA : aggB;
      hipMemsetAsync(acc, 0, (size_t)N * F * 4, stream);
      const int nt = E << 4;
      k_scatter<<<(nt + TB - 1) / TB, TB, 0, stream>>>(cur, srcI, dstI, acc, E);
      k_scale_convert<<<(nf + TB - 1) / TB, TB, 0, stream>>>(acc, invcnt, acc,
                                                             x16, nf);
      cur = acc;
    }
    // layer 0
    k_gru_step<<<gruBlocks, TB, 0, stream>>>(x16, wih0, whh0, b_ih0, b_hh0,
                                             h0_16, h0_32, h0_32, h0_16, N,
                                             t == 0);
    // layer 1 consumes layer-0 hidden directly (no [T,N,H] buffer needed)
    k_gru_step<<<gruBlocks, TB, 0, stream>>>(h0_16, wih1, whh1, b_ih1, b_hh1,
                                             h1_16, h1_32, h1_32, h1_16, N,
                                             t == 0);
  }

  k_head<<<(N + TB - 1) / TB, TB, 0, stream>>>(h1_32, W_out, b_out, out, N);
}
